// SRUpp__35845797053293
// MI455X (gfx1250) — compile-verified, matrix operand order checked by reading, not run on hardware
//
#include <hip/hip_runtime.h>
#include <hip/hip_bf16.h>
#include <math.h>

// ---------------- problem constants (from reference) ----------------
#define TT    1024
#define BB    16
#define DD    1024
#define PP    512
#define NROWS (TT * BB)       // 16384 flattened (t,b) rows

#define BLK_M 256             // rows per workgroup
#define BLK_N 64              // cols per workgroup (4 n-tiles)

typedef __attribute__((ext_vector_type(16))) __bf16 v16bf;
typedef __attribute__((ext_vector_type(8)))  __bf16 v8bf;
typedef __attribute__((ext_vector_type(8)))  float  v8f;
typedef __attribute__((ext_vector_type(4)))  unsigned int v4u;
typedef __attribute__((ext_vector_type(8)))  int    v8i;
typedef __attribute__((ext_vector_type(4)))  int    v4i;

__device__ __forceinline__ __bf16 f2bf(float f) { return (__bf16)f; }

// ---------------------------------------------------------------------------
// TDM: 2-D tile load Global -> LDS via Tensor Data Mover (D# per ISA ch.8).
//   d0 elements (data_size=2B) per line, d1 lines, line stride = strideElems.
//   group0: count=1 | lds_addr | global_addr(57b) | type=2
//   group1: data_size=1(2B), tensor_dim0=d0, tensor_dim1=d1, tile_dim0=d0,
//           tile_dim1=d1, tensor_dim0_stride=strideElems
// ---------------------------------------------------------------------------
__device__ __forceinline__ void tdm_load_2d(unsigned ldsOff, const void* gptr,
                                            unsigned d0, unsigned d1,
                                            unsigned strideElems)
{
    unsigned long long ga = (unsigned long long)gptr;
    v4u g0 = { 1u,                                   // count = 1 valid descriptor
               ldsOff,                               // lds_addr (bytes)
               (unsigned)(ga & 0xFFFFFFFFu),
               (unsigned)((ga >> 32) & 0x01FFFFFFu) | 0x80000000u };  // type=2
    v8i g1 = { (int)0x00010000u,                     // wg_mask=0, data_size=1 (2B)
               (int)((d0 & 0xFFFFu) << 16),          // tensor_dim0[15:0]
               (int)(((d0 >> 16) & 0xFFFFu) | ((d1 & 0xFFFFu) << 16)),
               (int)(((d1 >> 16) & 0xFFFFu) | ((d0 & 0xFFFFu) << 16)), // tile_dim0
               (int)(d1 & 0xFFFFu),                  // tile_dim1
               (int)strideElems,                     // tensor_dim0_stride[31:0]
               0, 0 };
    v4i g2 = { 0, 0, 0, 0 };
    v4i g3 = { 0, 0, 0, 0 };
#if defined(__clang_major__) && (__clang_major__ >= 23)
    v8i g4 = { 0, 0, 0, 0, 0, 0, 0, 0 };
    __builtin_amdgcn_tensor_load_to_lds(g0, g1, g2, g3, g4, 0);
#else
    __builtin_amdgcn_tensor_load_to_lds(g0, g1, g2, g3, 0);
#endif
}

// ---------------------------------------------------------------------------
// Pack f32 matrix (K x N, arbitrary strides, optional batch) into WMMA-B
// fragment order: dst[((kt*tilesN+nt)*32 + lane)*16 + e] = B[k(e), n(lane)].
// ---------------------------------------------------------------------------
__global__ __launch_bounds__(256)
void pack_b_frags(const float* __restrict__ src,
                  long long kStride, long long nStride, long long srcBatchStride,
                  __bf16* __restrict__ dst, long long dstBatchStride,
                  int K, int N)
{
    int tid    = blockIdx.x * blockDim.x + threadIdx.x;
    int tilesN = N >> 4;
    int total  = (K >> 5) * tilesN * 32;
    if (tid >= total) return;
    int lane = tid & 31;
    int fi   = tid >> 5;
    int nt   = fi % tilesN;
    int kt   = fi / tilesN;
    int half = lane >> 4;
    int n    = nt * 16 + (lane & 15);
    const float* s = src + (long long)blockIdx.z * srcBatchStride;
    __bf16*      d = dst + (long long)blockIdx.z * dstBatchStride + (long long)tid * 16;
    int kbase = kt * 32 + half * 8;
#pragma unroll
    for (int e = 0; e < 16; ++e) {
        int k = kbase + e + ((e < 8) ? 0 : 8);
        d[e] = f2bf(s[(long long)k * kStride + (long long)n * nStride]);
    }
}

// --------------------- bulk f32 -> bf16 conversion -------------------------
__global__ __launch_bounds__(256)
void f32_to_bf16(const float* __restrict__ src, __bf16* __restrict__ dst,
                 long long n8)            // n/8 work items
{
    long long i = (long long)blockIdx.x * blockDim.x + threadIdx.x;
    if (i >= n8) return;
    const float4* s = (const float4*)(src + i * 8);
    float4 a = s[0], b = s[1];
    v8bf o;
    o[0] = f2bf(a.x); o[1] = f2bf(a.y); o[2] = f2bf(a.z); o[3] = f2bf(a.w);
    o[4] = f2bf(b.x); o[5] = f2bf(b.y); o[6] = f2bf(b.z); o[7] = f2bf(b.w);
    *(v8bf*)(dst + i * 8) = o;
}

// ---------------------------------------------------------------------------
// Panel GEMM: block computes BLK_M x BLK_N, 8 waves, each wave 2x4 WMMA tiles.
// A (bf16 row-major) and packed-B panels are staged into LDS by the Tensor
// Data Mover (wave 0 issues, s_wait_tensorcnt + barrier publishes), double
// buffered. Epilogue: alpha*acc (+Add), optional f32 and/or bf16 outputs.
// ---------------------------------------------------------------------------
__global__ __launch_bounds__(256)
void wmma_gemm_tdm(const __bf16* __restrict__ A, long long aBase, int lda, long long aBatch,
                   const __bf16* __restrict__ Bp, long long bBatch, int Ntiles,
                   float* __restrict__ Cf, __bf16* __restrict__ Cb,
                   long long cBase, int ldc, long long cBatch,
                   const float* __restrict__ Add,
                   int K, float alpha)
{
    __shared__ __bf16 ldsA[2][BLK_M * 32];   // 2 x 16 KB
    __shared__ __bf16 ldsB[2][BLK_N * 32];   // 2 x  4 KB

    int tid  = threadIdx.x;
    int lane = tid & 31;
    int w    = tid >> 5;
    int half = lane >> 4;
    int lm   = lane & 15;

    int Nblk = Ntiles >> 2;                  // super-cols
    int bm   = blockIdx.x / Nblk;
    int bn   = blockIdx.x % Nblk;
    long long batch = blockIdx.z;

    const __bf16* Ab  = A  + aBase + batch * aBatch + (long long)(bm * BLK_M) * lda;
    const __bf16* Bpb = Bp + batch * bBatch;

    int KT = K >> 5;

    // preload panel 0
    if (tid < 32) {
        tdm_load_2d((unsigned)(size_t)(void*)&ldsA[0][0], Ab, 32, BLK_M, (unsigned)lda);
        tdm_load_2d((unsigned)(size_t)(void*)&ldsB[0][0],
                    Bpb + (long long)(0 * Ntiles + bn * 4) * 512,
                    BLK_N * 32, 1, BLK_N * 32);
    }

    v8f acc[2][4];
#pragma unroll
    for (int i = 0; i < 2; ++i)
#pragma unroll
        for (int j = 0; j < 4; ++j)
            acc[i][j] = (v8f){};

    for (int kt = 0; kt < KT; ++kt) {
        int buf = kt & 1;
        if (tid < 32) {
            if (kt + 1 < KT) {
                int nb = (kt + 1) & 1;
                tdm_load_2d((unsigned)(size_t)(void*)&ldsA[nb][0],
                            Ab + (kt + 1) * 32, 32, BLK_M, (unsigned)lda);
                tdm_load_2d((unsigned)(size_t)(void*)&ldsB[nb][0],
                            Bpb + (long long)((kt + 1) * Ntiles + bn * 4) * 512,
                            BLK_N * 32, 1, BLK_N * 32);
                __builtin_amdgcn_s_wait_tensorcnt(2);   // pair(kt) complete
            } else {
                __builtin_amdgcn_s_wait_tensorcnt(0);
            }
        }
        __syncthreads();

        // B fragments (shared by all waves): one 32B LDS read per n-tile
        v16bf bfr[4];
#pragma unroll
        for (int tn = 0; tn < 4; ++tn)
            bfr[tn] = *(const v16bf*)&ldsB[buf][(tn * 32 + lane) * 16];

#pragma unroll
        for (int tm = 0; tm < 2; ++tm) {
            int rowL = w * 32 + tm * 16 + lm;
            v8bf lo = *(const v8bf*)&ldsA[buf][rowL * 32 + half * 8];
            v8bf hi = *(const v8bf*)&ldsA[buf][rowL * 32 + 16 + half * 8];
            v16bf a = __builtin_shufflevector(lo, hi, 0, 1, 2, 3, 4, 5, 6, 7,
                                                      8, 9, 10, 11, 12, 13, 14, 15);
#pragma unroll
            for (int tn = 0; tn < 4; ++tn)
                acc[tm][tn] = __builtin_amdgcn_wmma_f32_16x16x32_bf16(
                    false, a, false, bfr[tn], (short)0, acc[tm][tn], false, false);
        }
        __syncthreads();
    }

    long long cb = cBase + batch * cBatch;
#pragma unroll
    for (int tm = 0; tm < 2; ++tm) {
#pragma unroll
        for (int tn = 0; tn < 4; ++tn) {
            int col = bn * BLK_N + tn * 16 + lm;
#pragma unroll
            for (int r = 0; r < 8; ++r) {
                int row = bm * BLK_M + w * 32 + tm * 16 + half * 8 + r;
                long long idx = cb + (long long)row * ldc + col;
                float v = alpha * acc[tm][tn][r];
                if (Add) v += Add[idx];
                if (Cf)  Cf[idx] = v;
                if (Cb)  Cb[idx] = f2bf(v);
            }
        }
    }
}

// ------------- causal softmax: f32 scores in, bf16 probs out ---------------
__global__ __launch_bounds__(256)
void causal_softmax(const float* __restrict__ S, __bf16* __restrict__ Pb)
{
    __shared__ float red[256];
    int row = blockIdx.x;                    // b*T + t
    int t   = row & (TT - 1);
    const float* p = S + (long long)row * TT;
    __bf16* o = Pb + (long long)row * TT;
    int valid = t + 1;
    int tid = threadIdx.x;

    float mx = -3.4e38f;
    for (int i = tid; i < valid; i += 256) mx = fmaxf(mx, p[i]);
    red[tid] = mx; __syncthreads();
    for (int s = 128; s > 0; s >>= 1) {
        if (tid < s) red[tid] = fmaxf(red[tid], red[tid + s]);
        __syncthreads();
    }
    mx = red[0]; __syncthreads();

    float sum = 0.f;
    for (int i = tid; i < valid; i += 256) sum += __expf(p[i] - mx);
    red[tid] = sum; __syncthreads();
    for (int s = 128; s > 0; s >>= 1) {
        if (tid < s) red[tid] += red[tid + s];
        __syncthreads();
    }
    float inv = 1.f / red[0];

    for (int i = tid; i < TT; i += 256)
        o[i] = f2bf((i < valid) ? __expf(p[i] - mx) * inv : 0.f);
}

// ------------------------- SRU recurrence (scan over T) --------------------
__global__ __launch_bounds__(256)
void sru_scan(const float* __restrict__ U, const float* __restrict__ X,
              const float* __restrict__ vf, const float* __restrict__ vr,
              const float* __restrict__ bf, const float* __restrict__ br,
              float* __restrict__ H)
{
    int tid = blockIdx.x * blockDim.x + threadIdx.x;     // one per (b,d)
    if (tid >= BB * DD) return;
    int b = tid / DD;
    int d = tid % DD;
    float vfd = vf[d], vrd = vr[d], bfd = bf[d], brd = br[d];
    float c = 0.f;
    for (int t = 0; t < TT; ++t) {
        long long r = (long long)t * BB + b;
        const float* ut = U + r * (3 * DD);
        float u0 = ut[d], u1 = ut[DD + d], u2 = ut[2 * DD + d];
        float fg = 1.f / (1.f + __expf(-(u1 + vfd * c + bfd)));
        float cn = fg * c + (1.f - fg) * u0;
        float rg = 1.f / (1.f + __expf(-(u2 + vrd * c + brd)));   // uses c_{t-1}
        H[r * DD + d] = rg * fmaxf(cn, 0.f) + (1.f - rg) * X[r * DD + d];
        c = cn;
    }
}

// ---------------------------------------------------------------------------
extern "C" void kernel_launch(void* const* d_in, const int* in_sizes, int n_in,
                              void* d_out, int out_size, void* d_ws, size_t ws_size,
                              hipStream_t stream)
{
    (void)in_sizes; (void)n_in; (void)out_size; (void)ws_size;
    const float* x = (const float*)d_in[0];

    char* ws = (char*)d_ws;
    size_t off = 0;
    auto alloc = [&](size_t bytes) -> void* {
        void* p = ws + off;
        off = (off + bytes + 255) & ~(size_t)255;
        return p;
    };
    __bf16* xh   = (__bf16*)alloc((size_t)NROWS * DD * 2);        // layer input, bf16
    float*  qf   = (float*) alloc((size_t)NROWS * PP * 4);        // q, f32 (residual)
    __bf16* qh   = (__bf16*)alloc((size_t)NROWS * PP * 2);        // q, bf16 (A side)
    float*  kf   = (float*) alloc((size_t)NROWS * PP * 4);
    float*  vf32 = (float*) alloc((size_t)NROWS * PP * 4);
    __bf16* sfh  = (__bf16*)alloc((size_t)NROWS * PP * 2);        // q + attn, bf16
    float*  sc   = (float*) alloc((size_t)BB * TT * TT * 4);      // scores
    __bf16* pb   = (__bf16*)alloc((size_t)BB * TT * TT * 2);      // probs, bf16
    float*  u    = (float*) alloc((size_t)NROWS * 3 * DD * 4);
    float*  h1   = (float*) alloc((size_t)NROWS * DD * 4);
    __bf16* pW   = (__bf16*)alloc((size_t)PP * 3 * DD * 2);       // weight pack (reused)
    __bf16* pKV  = (__bf16*)alloc((size_t)BB * TT * PP * 2);      // K/V pack (reused)

    const float inv_sqrt_p = 0.044194173824159216f;               // 1/sqrt(512)
    const long long kvFrag = (long long)(PP / 32) * (TT / 16) * 32 * 16;  // 524288

    for (int l = 0; l < 2; ++l) {
        const float* Wq = (const float*)d_in[1 + 8 * l + 0];
        const float* Wk = (const float*)d_in[1 + 8 * l + 1];
        const float* Wv = (const float*)d_in[1 + 8 * l + 2];
        const float* Wo = (const float*)d_in[1 + 8 * l + 3];
        const float* vf = (const float*)d_in[1 + 8 * l + 4];
        const float* vr = (const float*)d_in[1 + 8 * l + 5];
        const float* bf = (const float*)d_in[1 + 8 * l + 6];
        const float* br = (const float*)d_in[1 + 8 * l + 7];
        const float* X  = (l == 0) ? x : h1;
        float*       H  = (l == 0) ? h1 : (float*)d_out;

        // ---- X -> bf16
        f32_to_bf16<<<dim3((size_t)NROWS * DD / 8 / 256, 1, 1), 256, 0, stream>>>(
            X, xh, (long long)NROWS * DD / 8);

        // ---- q = X @ Wq  (M=16384, K=1024, N=512); outputs f32 + bf16
        pack_b_frags<<<dim3((DD / 32) * (PP / 16) * 32 / 256, 1, 1), 256, 0, stream>>>(
            Wq, PP, 1, 0, pW, 0, DD, PP);
        wmma_gemm_tdm<<<dim3((NROWS / BLK_M) * (PP / BLK_N), 1, 1), 256, 0, stream>>>(
            xh, 0, DD, 0, pW, 0, PP / 16, qf, qh, 0, PP, 0, nullptr, DD, 1.f);

        // ---- k = q @ Wk  (M=16384, K=512, N=512)
        pack_b_frags<<<dim3((PP / 32) * (PP / 16) * 32 / 256, 1, 1), 256, 0, stream>>>(
            Wk, PP, 1, 0, pW, 0, PP, PP);
        wmma_gemm_tdm<<<dim3((NROWS / BLK_M) * (PP / BLK_N), 1, 1), 256, 0, stream>>>(
            qh, 0, PP, 0, pW, 0, PP / 16, kf, nullptr, 0, PP, 0, nullptr, PP, 1.f);

        // ---- v = q @ Wv
        pack_b_frags<<<dim3((PP / 32) * (PP / 16) * 32 / 256, 1, 1), 256, 0, stream>>>(
            Wv, PP, 1, 0, pW, 0, PP, PP);
        wmma_gemm_tdm<<<dim3((NROWS / BLK_M) * (PP / BLK_N), 1, 1), 256, 0, stream>>>(
            qh, 0, PP, 0, pW, 0, PP / 16, vf32, nullptr, 0, PP, 0, nullptr, PP, 1.f);

        // ---- scores[b] = q_b @ k_b^T / sqrt(P); B[p,s] = kf[(s*B+b)*P + p]
        pack_b_frags<<<dim3((PP / 32) * (TT / 16) * 32 / 256, 1, BB), 256, 0, stream>>>(
            kf, 1, (long long)BB * PP, PP, pKV, kvFrag, PP, TT);
        wmma_gemm_tdm<<<dim3((TT / BLK_M) * (TT / BLK_N), 1, BB), 256, 0, stream>>>(
            qh, 0, BB * PP, PP, pKV, kvFrag, TT / 16,
            sc, nullptr, 0, TT, (long long)TT * TT, nullptr, PP, inv_sqrt_p);

        // ---- causal softmax -> bf16 probs
        causal_softmax<<<dim3(BB * TT, 1, 1), 256, 0, stream>>>(sc, pb);

        // ---- sf = attn @ v + q (bf16 out); B[s,p] = vf32[(s*B+b)*P + p]
        pack_b_frags<<<dim3((TT / 32) * (PP / 16) * 32 / 256, 1, BB), 256, 0, stream>>>(
            vf32, (long long)BB * PP, 1, PP, pKV, kvFrag, TT, PP);
        wmma_gemm_tdm<<<dim3((TT / BLK_M) * (PP / BLK_N), 1, BB), 256, 0, stream>>>(
            pb, 0, TT, (long long)TT * TT, pKV, kvFrag, PP / 16,
            nullptr, sfh, 0, BB * PP, PP, qf, TT, 1.f);

        // ---- u = (q + a) @ Wo  (M=16384, K=512, N=3072)
        pack_b_frags<<<dim3((PP / 32) * (3 * DD / 16) * 32 / 256, 1, 1), 256, 0, stream>>>(
            Wo, 3 * DD, 1, 0, pW, 0, PP, 3 * DD);
        wmma_gemm_tdm<<<dim3((NROWS / BLK_M) * (3 * DD / BLK_N), 1, 1), 256, 0, stream>>>(
            sfh, 0, PP, 0, pW, 0, 3 * DD / 16, u, nullptr, 0, 3 * DD, 0, nullptr, PP, 1.f);

        // ---- sequential SRU recurrence over T
        sru_scan<<<dim3(BB * DD / 256, 1, 1), 256, 0, stream>>>(u, X, vf, vr, bf, br, H);
    }
}
